// Graphormer_73615739453468
// MI455X (gfx1250) — matrix operand order repaired and checked by reading, not compiled, not run on hardware
//
#include <hip/hip_runtime.h>
#include <hip/hip_bf16.h>
#include <math.h>

#define N_NODES 2048
#define N_EDGES 65536
#define FDIM    128
#define HDIM    512
#define EFDIM   16
#define EDDIM   64
#define LPATH   5
#define NLAYER  4
#define NHEAD   8
#define DK      64
#define ODIM    64
#define MAXDEG  64

typedef __attribute__((ext_vector_type(16))) _Float16 v16h;
typedef __attribute__((ext_vector_type(8)))  _Float16 v8h;
typedef __attribute__((ext_vector_type(8)))  float    v8f;

// ---------------------------------------------------------------------------
// WMMA helpers (CDNA5 16x16x32 f16, wave32)
// A frag: lane row M=lane&15; elem i -> K = k0 + i + 16*(lane>=16)
// B frag: lane col N=lane&15; elem i -> K = k0 + i + 16*(lane>=16)
// C/D:    vgpr r, lane -> (M = r + 8*(lane>=16), N = lane&15)
// ---------------------------------------------------------------------------
__device__ __forceinline__ v8f wmma_f16(v16h a, v16h b, v8f c) {
  return __builtin_amdgcn_wmma_f32_16x16x32_f16(false, a, false, b, (short)0, c,
                                                false, false);
}

// 16 contiguous halves -> fragment (elem i maps to offset i)
__device__ __forceinline__ v16h frag_contig(const _Float16* p) {
  v8h lo = *(const v8h*)p, hi = *(const v8h*)(p + 8);
  v16h a;
#pragma unroll
  for (int i = 0; i < 8; ++i) { a[i] = lo[i]; a[i + 8] = hi[i]; }
  return a;
}

// A fragment: 16 contiguous halves (elem i maps to offset i)
__device__ __forceinline__ v16h frag_a_row(const _Float16* row) {
  v8h lo = *(const v8h*)row, hi = *(const v8h*)(row + 8);
  v16h a;
#pragma unroll
  for (int i = 0; i < 8; ++i) { a[i] = lo[i]; a[i + 8] = hi[i]; }
  return a;
}

__device__ __forceinline__ v16h load_a_frag_h(const _Float16* A, int lda, int m0,
                                              int k0, int lane) {
  return frag_a_row(A + (size_t)(m0 + (lane & 15)) * lda + k0 + (lane & 16));
}

__device__ __forceinline__ float gelu_tanh(float x) {
  float x3 = x * x * x;
  return 0.5f * x * (1.0f + tanhf(0.7978845608028654f * (x + 0.044715f * x3)));
}

// ---------------------------------------------------------------------------
// f32 -> f16 bulk convert (count multiple of 8)
// ---------------------------------------------------------------------------
__global__ void cvt_f16_kernel(const float* __restrict__ src,
                               _Float16* __restrict__ dst, int n8) {
  int i = blockIdx.x * blockDim.x + threadIdx.x;
  if (i >= n8) return;
  const float4* s4 = (const float4*)src;
  float4 a = s4[2 * i], b = s4[2 * i + 1];
  v8h d;
  d[0] = (_Float16)a.x; d[1] = (_Float16)a.y; d[2] = (_Float16)a.z; d[3] = (_Float16)a.w;
  d[4] = (_Float16)b.x; d[5] = (_Float16)b.y; d[6] = (_Float16)b.z; d[7] = (_Float16)b.w;
  ((v8h*)dst)[i] = d;
}

// ---------------------------------------------------------------------------
// Degrees
// ---------------------------------------------------------------------------
__global__ void deg_kernel(const int* __restrict__ edge_index,
                           int* __restrict__ in_deg, int* __restrict__ out_deg) {
  int e = blockIdx.x * blockDim.x + threadIdx.x;
  if (e < N_EDGES) {
    atomicAdd(&out_deg[edge_index[e]], 1);
    atomicAdd(&in_deg[edge_index[N_EDGES + e]], 1);
  }
}

// ---------------------------------------------------------------------------
// w[e][l] = ((edge_attr[e] @ W_edge) + b_edge) . edge_vector[l]
// ---------------------------------------------------------------------------
__global__ void edge_w_kernel(const float* __restrict__ edge_attr,
                              const float* __restrict__ W_edge,
                              const float* __restrict__ b_edge,
                              const float* __restrict__ edge_vector,
                              float* __restrict__ w) {
  int e = blockIdx.x * blockDim.x + threadIdx.x;
  if (e >= N_EDGES) return;
  float a[EFDIM];
#pragma unroll
  for (int i = 0; i < EFDIM; ++i) a[i] = edge_attr[(size_t)e * EFDIM + i];
  float acc[LPATH] = {0.f, 0.f, 0.f, 0.f, 0.f};
  for (int d = 0; d < EDDIM; ++d) {
    float emb = b_edge[d];
#pragma unroll
    for (int i = 0; i < EFDIM; ++i) emb += a[i] * W_edge[i * EDDIM + d];
#pragma unroll
    for (int l = 0; l < LPATH; ++l) acc[l] += emb * edge_vector[l * EDDIM + d];
  }
#pragma unroll
  for (int l = 0; l < LPATH; ++l) w[(size_t)e * LPATH + l] = acc[l];
}

// ---------------------------------------------------------------------------
// bias[i][j]
// ---------------------------------------------------------------------------
__global__ void bias_kernel(const int* __restrict__ node_paths,
                            const int* __restrict__ edge_paths,
                            const float* __restrict__ w,
                            const float* __restrict__ b_spatial,
                            float* __restrict__ bias) {
  long long idx = (long long)blockIdx.x * blockDim.x + threadIdx.x;
  if (idx >= (long long)N_NODES * N_NODES) return;
  const int* np = node_paths + idx * LPATH;
  const int* ep = edge_paths + idx * LPATH;
  int plen = 0, cnt = 0;
  float csum = 0.f;
#pragma unroll
  for (int l = 0; l < LPATH; ++l) {
    if (np[l] >= 0) plen++;
    int e = ep[l];
    if (e >= 0) { csum += w[(size_t)e * LPATH + l]; cnt++; }
  }
  float c = (cnt > 0) ? csum / (float)cnt : 0.f;
  int pi = plen - 1;
  if (pi > LPATH - 1) pi = LPATH - 1;
  float bsp = (plen > 0) ? b_spatial[pi] : 0.f;
  bias[idx] = bsp + c;
}

// ---------------------------------------------------------------------------
// Block-cooperative WMMA GEMM: BMxBN tile, f16 A/B staged through LDS.
// A-panel row-major (stride 40 halves), B-panel transposed (Bs[col][k],
// stride 40 halves) -> every fragment read is two contiguous ds_load_b128.
// EPI: 0 plain->f16  1 +res->f32  2 gelu->f16  3 node-embed->f32  4 plain->f32
// ---------------------------------------------------------------------------
template <int BM, int BN, int EPI>
__global__ void __launch_bounds__((BM / 16) * (BN / 64) * 32, 1)
gemm_f16(const _Float16* __restrict__ A, const _Float16* __restrict__ B,
         const float* __restrict__ bias, const float* __restrict__ res,
         const float* __restrict__ z_in, const float* __restrict__ z_out,
         const int* __restrict__ in_deg, const int* __restrict__ out_deg,
         _Float16* __restrict__ Dh, float* __restrict__ Df,
         int M, int N, int K) {
  constexpr int NW = (BM / 16) * (BN / 64);
  constexpr int NTHR = NW * 32;
  constexpr int PAD = 40;  // halves: stride 20 words -> 16 distinct banks
  __shared__ _Float16 As[BM * PAD];
  __shared__ _Float16 Bs[BN * PAD];
  int tid = threadIdx.x;
  int lane = tid & 31;
  int w = tid >> 5;
  int wm = w & (BM / 16 - 1);
  int wn = w / (BM / 16);
  int mtiles = M / BM;
  int m0 = (blockIdx.x % mtiles) * BM;
  int n0 = (blockIdx.x / mtiles) * BN;

  v8f acc[4] = {};
  for (int k0 = 0; k0 < K; k0 += 32) {
    __syncthreads();
    // A panel: BM x 32 halves
    for (int idx = tid; idx < BM * 4; idx += NTHR) {
      int row = idx >> 2, seg = idx & 3;
      *(v8h*)(As + row * PAD + seg * 8) =
          *(const v8h*)(A + (size_t)(m0 + row) * K + k0 + seg * 8);
    }
    // B panel transposed: Bs[col][k]
    for (int idx = tid; idx < (BN / 16) * 32; idx += NTHR) {
      int row = idx / (BN / 16);
      int cseg = (idx % (BN / 16)) * 16;
      const _Float16* gp = B + (size_t)(k0 + row) * N + n0 + cseg;
      v8h d0 = *(const v8h*)gp;
      v8h d1 = *(const v8h*)(gp + 8);
#pragma unroll
      for (int j = 0; j < 8; ++j) {
        Bs[(cseg + j) * PAD + row] = d0[j];
        Bs[(cseg + 8 + j) * PAD + row] = d1[j];
      }
    }
    __syncthreads();
    v16h af = frag_a_row(As + (wm * 16 + (lane & 15)) * PAD + (lane & 16));
#pragma unroll
    for (int t = 0; t < 4; ++t) {
      v16h bf = frag_contig(Bs + (wn * 64 + t * 16 + (lane & 15)) * PAD + (lane & 16));
      acc[t] = wmma_f16(af, bf, acc[t]);
    }
  }

  int colb = n0 + wn * 64 + (lane & 15);
  int rowb = m0 + wm * 16 + ((lane & 16) >> 1);
#pragma unroll
  for (int t = 0; t < 4; ++t) {
    int col = colb + t * 16;
    float bvv = bias[col];
#pragma unroll
    for (int r = 0; r < 8; ++r) {
      int row = rowb + r;
      size_t idx = (size_t)row * N + col;
      float vv = acc[t][r] + bvv;
      if (EPI == 1) {
        Df[idx] = res[idx] + vv;
      } else if (EPI == 2) {
        Dh[idx] = (_Float16)gelu_tanh(vv);
      } else if (EPI == 3) {
        int di = in_deg[row];  if (di > MAXDEG - 1) di = MAXDEG - 1;
        int dq = out_deg[row]; if (dq > MAXDEG - 1) dq = MAXDEG - 1;
        Df[idx] = vv + z_in[(size_t)di * N + col] + z_out[(size_t)dq * N + col];
      } else if (EPI == 4) {
        Df[idx] = vv;
      } else {
        Dh[idx] = (_Float16)vv;
      }
    }
  }
}

// ---------------------------------------------------------------------------
// LayerNorm: one wave per row; f32 in, f16 out
// ---------------------------------------------------------------------------
__global__ void ln_kernel(const float* __restrict__ h, const float* __restrict__ s,
                          const float* __restrict__ b, _Float16* __restrict__ y) {
  int row = (blockIdx.x * blockDim.x + threadIdx.x) >> 5;
  int lane = threadIdx.x & 31;
  if (row >= N_NODES) return;
  const float* hr = h + (size_t)row * HDIM;
  float vals[16], sum = 0.f, sq = 0.f;
#pragma unroll
  for (int i = 0; i < 16; ++i) {
    float xv = hr[lane + 32 * i];
    vals[i] = xv; sum += xv; sq += xv * xv;
  }
#pragma unroll
  for (int m = 1; m < 32; m <<= 1) {
    sum += __shfl_xor(sum, m);
    sq  += __shfl_xor(sq, m);
  }
  float mean = sum * (1.0f / HDIM);
  float var  = sq * (1.0f / HDIM) - mean * mean;
  float inv  = rsqrtf(var + 1e-5f);
#pragma unroll
  for (int i = 0; i < 16; ++i) {
    int c = lane + 32 * i;
    y[(size_t)row * HDIM + c] = (_Float16)((vals[i] - mean) * inv * s[c] + b[c]);
  }
}

// ---------------------------------------------------------------------------
// Flash attention. Block = 8 waves, all same head, 8 consecutive query tiles.
// K chunk (32x64) and V^T chunk staged in LDS, shared by all waves. Online
// softmax with a single fused per-row pass. __launch_bounds__(256,1): the
// grid is only 128 blocks (launch-limited occupancy), so give the compiler
// the full VGPR allowance to keep Q fragments resident (no scratch spills).
// ---------------------------------------------------------------------------
__global__ void __launch_bounds__(256, 1)
attn_kernel(const _Float16* __restrict__ q, const _Float16* __restrict__ k,
            const _Float16* __restrict__ v, const float* __restrict__ bias,
            _Float16* __restrict__ out) {
  __shared__ _Float16 Ks[32 * 72];   // K rows, padded stride 72 halves
  __shared__ _Float16 Vt[64 * 40];   // V transposed: Vt[col][k], stride 40
  __shared__ _Float16 plds[8 * 512]; // one 16x32 P tile per wave
  int tid = threadIdx.x, lane = tid & 31, w = tid >> 5;
  int head = blockIdx.x >> 4;
  int qt = (blockIdx.x & 15) * 8 + w;
  int cbase = head * DK;
  int qrowb = qt * 16 + ((lane & 16) >> 1);
  int jrow = lane & 15, koff = lane & 16;

  v16h qa0 = load_a_frag_h(q, HDIM, qt * 16, cbase, lane);
  v16h qa1 = load_a_frag_h(q, HDIM, qt * 16, cbase + 32, lane);

  v8f o0 = {}, o1 = {}, o2 = {}, o3 = {};
  float mrow[8], srow[8];
#pragma unroll
  for (int r = 0; r < 8; ++r) { mrow[r] = -1e30f; srow[r] = 0.f; }

  _Float16* pw = plds + w * 512;
  int srow16 = tid >> 3, sseg = tid & 7;  // staging assignment (256 threads)
  const float* bbase = bias + (size_t)qrowb * N_NODES + jrow;  // + r*N + jb

#pragma unroll 1
  for (int jb = 0; jb < N_NODES; jb += 32) {
    __syncthreads();
    {  // stage K rows + transposed V
      const _Float16* kg = k + (size_t)(jb + srow16) * HDIM + cbase + sseg * 8;
      const _Float16* vg = v + (size_t)(jb + srow16) * HDIM + cbase + sseg * 8;
      *(v8h*)(Ks + srow16 * 72 + sseg * 8) = *(const v8h*)kg;
      v8h vd = *(const v8h*)vg;
#pragma unroll
      for (int j = 0; j < 8; ++j) Vt[(sseg * 8 + j) * 40 + srow16] = vd[j];
    }
    if (jb + 32 < N_NODES) {  // gfx1250 global_prefetch for next chunk
      __builtin_prefetch(k + (size_t)(jb + 32 + srow16) * HDIM + cbase, 0, 0);
      __builtin_prefetch(v + (size_t)(jb + 32 + srow16) * HDIM + cbase, 0, 0);
    }
    __syncthreads();

    // S tiles (16 queries x 32 keys), dk=64 in two chunks
    v8f st0 = {}, st1 = {};
    st0 = wmma_f16(qa0, frag_contig(Ks + jrow * 72 + koff), st0);
    st0 = wmma_f16(qa1, frag_contig(Ks + jrow * 72 + 32 + koff), st0);
    st1 = wmma_f16(qa0, frag_contig(Ks + (16 + jrow) * 72 + koff), st1);
    st1 = wmma_f16(qa1, frag_contig(Ks + (16 + jrow) * 72 + 32 + koff), st1);

    // fused bias + online softmax, one row at a time (minimal live state)
#pragma unroll
    for (int r = 0; r < 8; ++r) {
      const float* bp = bbase + (size_t)r * N_NODES + jb;
      float sv0 = st0[r] * 0.125f + bp[0];
      float sv1 = st1[r] * 0.125f + bp[16];
      float tm = fmaxf(sv0, sv1);
#pragma unroll
      for (int msk = 1; msk <= 8; msk <<= 1) tm = fmaxf(tm, __shfl_xor(tm, msk));
      float mnew = fmaxf(mrow[r], tm);
      float corr = __expf(mrow[r] - mnew);
      float p0 = __expf(sv0 - mnew);
      float p1 = __expf(sv1 - mnew);
      float ts = p0 + p1;
#pragma unroll
      for (int msk = 1; msk <= 8; msk <<= 1) ts += __shfl_xor(ts, msk);
      srow[r] = srow[r] * corr + ts;
      mrow[r] = mnew;
      o0[r] *= corr; o1[r] *= corr; o2[r] *= corr; o3[r] *= corr;
      int rr = r + (koff >> 1);
      pw[rr * 32 + jrow]      = (_Float16)p0;
      pw[rr * 32 + 16 + jrow] = (_Float16)p1;
    }
    // O += P (16x32) @ V (32x64)
    v16h pa = frag_a_row(pw + jrow * 32 + koff);
    o0 = wmma_f16(pa, frag_contig(Vt + (jrow)      * 40 + koff), o0);
    o1 = wmma_f16(pa, frag_contig(Vt + (16 + jrow) * 40 + koff), o1);
    o2 = wmma_f16(pa, frag_contig(Vt + (32 + jrow) * 40 + koff), o2);
    o3 = wmma_f16(pa, frag_contig(Vt + (48 + jrow) * 40 + koff), o3);
  }

  int col = lane & 15;
#pragma unroll
  for (int r = 0; r < 8; ++r) {
    float inv = 1.0f / srow[r];
    size_t base = (size_t)(qrowb + r) * HDIM + cbase;
    out[base + col]      = (_Float16)(o0[r] * inv);
    out[base + 16 + col] = (_Float16)(o1[r] * inv);
    out[base + 32 + col] = (_Float16)(o2[r] * inv);
    out[base + 48 + col] = (_Float16)(o3[r] * inv);
  }
}

// ---------------------------------------------------------------------------
// Host launcher
// ---------------------------------------------------------------------------
extern "C" void kernel_launch(void* const* d_in, const int* in_sizes, int n_in,
                              void* d_out, int out_size, void* d_ws, size_t ws_size,
                              hipStream_t stream) {
  const float* x          = (const float*)d_in[0];
  const int*   edge_index = (const int*)d_in[1];
  const float* edge_attr  = (const float*)d_in[2];
  const int*   node_paths = (const int*)d_in[3];
  const int*   edge_paths = (const int*)d_in[4];
  const float* W_node     = (const float*)d_in[5];
  const float* b_node     = (const float*)d_in[6];
  const float* W_edge     = (const float*)d_in[7];
  const float* b_edge     = (const float*)d_in[8];
  const float* z_in       = (const float*)d_in[9];
  const float* z_out      = (const float*)d_in[10];
  const float* b_spatial  = (const float*)d_in[11];
  const float* edge_vec   = (const float*)d_in[12];
  const float* ln1_s      = (const float*)d_in[13];
  const float* ln1_b      = (const float*)d_in[14];
  const float* Wq         = (const float*)d_in[15];
  const float* bq         = (const float*)d_in[16];
  const float* Wk         = (const float*)d_in[17];
  const float* bk         = (const float*)d_in[18];
  const float* Wv         = (const float*)d_in[19];
  const float* bvp        = (const float*)d_in[20];
  const float* Wo         = (const float*)d_in[21];
  const float* bo         = (const float*)d_in[22];
  const float* ln2_s      = (const float*)d_in[23];
  const float* ln2_b      = (const float*)d_in[24];
  const float* W1         = (const float*)d_in[25];
  const float* b1         = (const float*)d_in[26];
  const float* W2         = (const float*)d_in[27];
  const float* b2         = (const float*)d_in[28];
  const float* W_out      = (const float*)d_in[29];
  const float* b_out      = (const float*)d_in[30];

  char* ws = (char*)d_ws;
  size_t off = 0;
  auto take = [&](size_t bytes) -> char* {
    char* p = ws + off;
    off = (off + bytes + 255) & ~(size_t)255;
    return p;
  };
  const size_t NH_ = (size_t)N_NODES * HDIM;
  int*      degs   = (int*)take(2 * N_NODES * sizeof(int));
  int*      in_deg = degs;
  int*      outdeg = degs + N_NODES;
  float*    w_el   = (float*)take((size_t)N_EDGES * LPATH * sizeof(float));
  float*    biasM  = (float*)take((size_t)N_NODES * N_NODES * sizeof(float));
  float*    h      = (float*)take(NH_ * sizeof(float));
  _Float16* xh     = (_Float16*)take((size_t)N_NODES * FDIM * 2);
  _Float16* yh     = (_Float16*)take(NH_ * 2);
  _Float16* qh     = (_Float16*)take(NH_ * 2);
  _Float16* kh     = (_Float16*)take(NH_ * 2);
  _Float16* vh     = (_Float16*)take(NH_ * 2);
  _Float16* oh     = (_Float16*)take(NH_ * 2);
  _Float16* th     = (_Float16*)take(NH_ * 2);
  _Float16* hh     = (_Float16*)take(NH_ * 2);
  _Float16* wn_h   = (_Float16*)take((size_t)FDIM * HDIM * 2);
  _Float16* wq_h   = (_Float16*)take((size_t)NLAYER * HDIM * HDIM * 2);
  _Float16* wk_h   = (_Float16*)take((size_t)NLAYER * HDIM * HDIM * 2);
  _Float16* wv_h   = (_Float16*)take((size_t)NLAYER * HDIM * HDIM * 2);
  _Float16* wo_h   = (_Float16*)take((size_t)NLAYER * HDIM * HDIM * 2);
  _Float16* w1_h   = (_Float16*)take((size_t)NLAYER * HDIM * HDIM * 2);
  _Float16* w2_h   = (_Float16*)take((size_t)NLAYER * HDIM * HDIM * 2);
  _Float16* wout_h = (_Float16*)take((size_t)HDIM * ODIM * 2);

  auto cvt = [&](const float* s, _Float16* d, size_t n) {
    int n8 = (int)(n / 8);
    cvt_f16_kernel<<<(n8 + 255) / 256, 256, 0, stream>>>(s, d, n8);
  };

  // ---- one-time conversions ----
  cvt(x, xh, (size_t)N_NODES * FDIM);
  cvt(W_node, wn_h, (size_t)FDIM * HDIM);
  cvt(Wq, wq_h, (size_t)NLAYER * HDIM * HDIM);
  cvt(Wk, wk_h, (size_t)NLAYER * HDIM * HDIM);
  cvt(Wv, wv_h, (size_t)NLAYER * HDIM * HDIM);
  cvt(Wo, wo_h, (size_t)NLAYER * HDIM * HDIM);
  cvt(W1, w1_h, (size_t)NLAYER * HDIM * HDIM);
  cvt(W2, w2_h, (size_t)NLAYER * HDIM * HDIM);
  cvt(W_out, wout_h, (size_t)HDIM * ODIM);

  // ---- graph-structure precompute ----
  hipMemsetAsync(degs, 0, 2 * N_NODES * sizeof(int), stream);
  deg_kernel<<<N_EDGES / 256, 256, 0, stream>>>(edge_index, in_deg, outdeg);
  edge_w_kernel<<<N_EDGES / 256, 256, 0, stream>>>(edge_attr, W_edge, b_edge,
                                                   edge_vec, w_el);
  bias_kernel<<<(N_NODES * N_NODES) / 256, 256, 0, stream>>>(
      node_paths, edge_paths, w_el, b_spatial, biasM);

  // ---- node embedding: h = x@W_node + b + z_in[deg] + z_out[deg] ----
  gemm_f16<64, 128, 3><<<(N_NODES / 64) * (HDIM / 128), 256, 0, stream>>>(
      xh, wn_h, b_node, nullptr, z_in, z_out, in_deg, outdeg, nullptr, h,
      N_NODES, HDIM, FDIM);

  const int gblk = (N_NODES / 64) * (HDIM / 128);  // 128 blocks
  for (int l = 0; l < NLAYER; ++l) {
    size_t wo_ = (size_t)l * HDIM * HDIM;
    size_t bo_ = (size_t)l * HDIM;
    ln_kernel<<<N_NODES / 8, 256, 0, stream>>>(h, ln1_s + bo_, ln1_b + bo_, yh);
    gemm_f16<64, 128, 0><<<gblk, 256, 0, stream>>>(
        yh, wq_h + wo_, bq + bo_, nullptr, nullptr, nullptr, nullptr, nullptr,
        qh, nullptr, N_NODES, HDIM, HDIM);
    gemm_f16<64, 128, 0><<<gblk, 256, 0, stream>>>(
        yh, wk_h + wo_, bk + bo_, nullptr, nullptr, nullptr, nullptr, nullptr,
        kh, nullptr, N_NODES, HDIM, HDIM);
    gemm_f16<64, 128, 0><<<gblk, 256, 0, stream>>>(
        yh, wv_h + wo_, bvp + bo_, nullptr, nullptr, nullptr, nullptr, nullptr,
        vh, nullptr, N_NODES, HDIM, HDIM);
    attn_kernel<<<NHEAD * 16, 256, 0, stream>>>(qh, kh, vh, biasM, oh);
    gemm_f16<64, 128, 1><<<gblk, 256, 0, stream>>>(
        oh, wo_h + wo_, bo + bo_, h, nullptr, nullptr, nullptr, nullptr,
        nullptr, h, N_NODES, HDIM, HDIM);
    ln_kernel<<<N_NODES / 8, 256, 0, stream>>>(h, ln2_s + bo_, ln2_b + bo_, yh);
    gemm_f16<64, 128, 2><<<gblk, 256, 0, stream>>>(
        yh, w1_h + wo_, b1 + bo_, nullptr, nullptr, nullptr, nullptr, nullptr,
        th, nullptr, N_NODES, HDIM, HDIM);
    gemm_f16<64, 128, 1><<<gblk, 256, 0, stream>>>(
        th, w2_h + wo_, b2 + bo_, h, nullptr, nullptr, nullptr, nullptr,
        nullptr, h, N_NODES, HDIM, HDIM);
  }

  // ---- output projection ----
  cvt(h, hh, NH_);
  gemm_f16<64, 64, 4><<<(N_NODES / 64) * (ODIM / 64), 128, 0, stream>>>(
      hh, wout_h, b_out, nullptr, nullptr, nullptr, nullptr, nullptr,
      nullptr, (float*)d_out, N_NODES, ODIM, HDIM);
}
